// MattNet_11948599018109
// MI455X (gfx1250) — compile-verified, matrix operand order checked by reading, not run on hardware
//
#include <hip/hip_runtime.h>
#include <math.h>

// ---------------------------------------------------------------------------
// Problem constants (match reference)
// ---------------------------------------------------------------------------
#define NNODES   8192
#define NEDGES   131072
#define CDIM     64
#define MTOT     9
#define NBASIS   64
#define NHEADS   8
#define DHEAD    8
#define MAXZ     100
#define RCUT     5.0f
#define NORMC    0.12649110641f     // sqrt(2/5^3)

typedef __attribute__((ext_vector_type(16))) _Float16 v16h;
typedef __attribute__((ext_vector_type(8)))  float    v8f;

// Async global->LDS staging (CDNA5) if this toolchain declares the builtins.
// Probe round 2 showed the builtin exists with param0:
//   'int __attribute__((vector_size(16))) __device__ *'  (AS1 pointer to int4)
#if defined(__HIP_DEVICE_COMPILE__) && \
    __has_builtin(__builtin_amdgcn_global_load_async_to_lds_b128) && \
    __has_builtin(__builtin_amdgcn_s_wait_asynccnt)
#define HAVE_ASYNC_LDS 1
typedef int v4i_ __attribute__((vector_size(16)));
typedef __attribute__((address_space(1))) v4i_* gv4ip_t;
typedef __attribute__((address_space(3))) v4i_* lv4ip_t;
#else
#define HAVE_ASYNC_LDS 0
#endif

// ---------------------------------------------------------------------------
// Workspace layout (floats), then an f16 weight arena
// ---------------------------------------------------------------------------
#define NMC      ((size_t)NNODES * MTOT * CDIM)               // 4,718,592
#define OFF_X    ((size_t)0)
#define OFF_H    (1ull * NMC)            // norm output / attention accumulator
#define OFF_Q    (2ull * NMC)
#define OFF_K    (3ull * NMC)
#define OFF_V    (4ull * NMC)
#define OFF_FFN  (2ull * NMC)            // overlays q,k,v (+1 extra NMC) after attention
#define OFF_LOG  (6ull * NMC)                                   // E*H
#define OFF_MX   (OFF_LOG + (size_t)NEDGES * NHEADS)            // N*H
#define OFF_DEN  (OFF_MX  + (size_t)NNODES * NHEADS)            // N*H
#define OFF_ENV  (OFF_DEN + (size_t)NNODES * NHEADS)            // E
#define OFF_SUM  (OFF_ENV + (size_t)NEDGES)                     // N*NB
#define OFF_EQ9  (OFF_SUM + (size_t)NNODES * NBASIS)            // N*9
#define OFF_INV  (OFF_EQ9 + (size_t)NNODES * MTOT)              // N*128
#define FLT_END  (OFF_INV + (size_t)NNODES * 2 * CDIM)

// f16 arena offsets (halfs) -- swizzled fragment-order weights
#define AOFF_RAD   ((size_t)0)                                  // 64*128
#define ARAD_SZ    (64 * 128)
#define ABLK_SZ    (4 * 3 * 64 * 64 + 2 * 3 * 64 * 256)         // 147456
#define AOFF_BLK(b) (ARAD_SZ + (size_t)(b) * ABLK_SZ)
#define AW_Q   0
#define AW_K   12288
#define AW_V   24576
#define AW_O   36864
#define AW_1   49152
#define AW_2   98304

// ---------------------------------------------------------------------------
// Utility kernels
// ---------------------------------------------------------------------------
__global__ void fill_kernel(float* __restrict__ p, float v, int n) {
    int i = blockIdx.x * blockDim.x + threadIdx.x;
    if (i < n) p[i] = v;
}

__global__ void copy_kernel(const float* __restrict__ s, float* __restrict__ d, int n) {
    int i = blockIdx.x * blockDim.x + threadIdx.x;
    if (i < n) d[i] = s[i];
}

// ---------------------------------------------------------------------------
// Weight swizzle: f32 [nmat][Cin][Cout] row-major  ->  f16 fragment order:
//   [mat][colgroup(Cout/64)][kstep(Cin/32)][tile(4)][lane(32)][16 halfs]
// Element (lane, j) of a fragment holds W[k][col] with
//   kloc = (lane>=16)?8:0 ;  k = kstep*32 + kloc + (j<8 ? j : 16 + (j-8))
//   col  = cg*64 + tile*16 + (lane&15)
// so a lane's whole B fragment is one contiguous, 32B-aligned 16-half run.
// ---------------------------------------------------------------------------
__global__ void swizzle_w_kernel(const float* __restrict__ src, _Float16* __restrict__ dst,
                                 int Cin, int Cout, int nmat)
{
    int idx = blockIdx.x * blockDim.x + threadIdx.x;
    const int per = Cin * Cout;
    if (idx >= nmat * per) return;
    const int l  = idx / per;
    const int r  = idx % per;
    const int slab = Cin * 64;
    const int cg = r / slab;
    const int r2 = r % slab;
    const int j    = r2 & 15;
    const int lane = (r2 >> 4) & 31;
    const int tt   = (r2 >> 9) & 3;
    const int ks   = r2 >> 11;
    const int kloc = (lane >= 16) ? 8 : 0;
    const int k    = ks * 32 + kloc + ((j < 8) ? j : (16 + (j - 8)));
    const int col  = cg * 64 + tt * 16 + (lane & 15);
    dst[idx] = (_Float16)src[(size_t)l * per + (size_t)k * Cout + col];
}

__device__ __forceinline__ void atomicMaxF(float* addr, float val) {
    int* ia = (int*)addr;
    int old = *ia;
    while (__int_as_float(old) < val) {
        int assumed = old;
        old = atomicCAS(ia, assumed, __float_as_int(val));
        if (old == assumed) break;
    }
}

__device__ __forceinline__ float sigmoidf_(float x) { return 1.0f / (1.0f + __expf(-x)); }

// ---------------------------------------------------------------------------
// WMMA GEMM:  Out[r, :] (+)= A[r, :] * W  (+ bias)
// W is pre-swizzled (see swizzle_w_kernel); the column-group slab for
// blockIdx.y is contiguous (Cin*64 halfs) and staged to LDS, async if
// available. Each lane then reads its B fragment as one aligned v16h.
// Block = 128 threads (4 waves); wave w owns rows [blk*64 + w*16, +16).
// ---------------------------------------------------------------------------
template <int CIN>
__global__ void __launch_bounds__(128) so3_gemm_kernel(
    const float* __restrict__ A, const _Float16* __restrict__ W,
    const float* __restrict__ bias, float* __restrict__ Out,
    int Cout, int mstart, int mcount, int strided, int accum)
{
    __shared__ __align__(32) _Float16 sW[CIN * 64];

    const int colbase = blockIdx.y * 64;
    const _Float16* __restrict__ slab = W + (size_t)blockIdx.y * (CIN * 64);

#if HAVE_ASYNC_LDS
    {
        const unsigned long long g = (unsigned long long)(const void*)slab;
        char* l = (char*)sW;
        for (int i = threadIdx.x; i < CIN * 8; i += 128)   // 16B chunks
            __builtin_amdgcn_global_load_async_to_lds_b128(
                (gv4ip_t)(g + (unsigned long long)i * 16),
                (lv4ip_t)(l + i * 16), 0, 0);
        __builtin_amdgcn_s_wait_asynccnt(0);
    }
#else
    for (int i = threadIdx.x; i < CIN * 8; i += 128)
        ((float4*)sW)[i] = ((const float4*)slab)[i];
#endif
    __syncthreads();

    const int wave  = threadIdx.x >> 5;
    const int lane  = threadIdx.x & 31;
    const int lr    = lane & 15;
    const int hi    = lane >> 4;       // 0/1 lane half
    const int kloc  = hi << 3;         // 0 or 8
    const int rbase = blockIdx.x * 64 + wave * 16;

    const int r    = rbase + lr;
    const int mrow = strided ? ((r / mcount) * MTOT + mstart + (r % mcount)) : r;
    const float* __restrict__ arow = A + (size_t)mrow * CIN;

    v8f acc[4] = {v8f{}, v8f{}, v8f{}, v8f{}};

#pragma unroll
    for (int kk = 0; kk < CIN; kk += 32) {
        if (kk + 32 < CIN) __builtin_prefetch(arow + kk + 32, 0, 1);
        // --- A fragment: 16 f32 -> f16 ---
        const float4 p0 = *(const float4*)(arow + kk + kloc);
        const float4 p1 = *(const float4*)(arow + kk + kloc + 4);
        const float4 p2 = *(const float4*)(arow + kk + kloc + 16);
        const float4 p3 = *(const float4*)(arow + kk + kloc + 20);
        v16h a;
        a[0]  = (_Float16)p0.x; a[1]  = (_Float16)p0.y; a[2]  = (_Float16)p0.z; a[3]  = (_Float16)p0.w;
        a[4]  = (_Float16)p1.x; a[5]  = (_Float16)p1.y; a[6]  = (_Float16)p1.z; a[7]  = (_Float16)p1.w;
        a[8]  = (_Float16)p2.x; a[9]  = (_Float16)p2.y; a[10] = (_Float16)p2.z; a[11] = (_Float16)p2.w;
        a[12] = (_Float16)p3.x; a[13] = (_Float16)p3.y; a[14] = (_Float16)p3.z; a[15] = (_Float16)p3.w;

        const int ks = kk >> 5;
#pragma unroll
        for (int t = 0; t < 4; ++t) {
            const v16h b = *(const v16h*)(sW + ((((ks * 4 + t) * 32) + lane) << 4));
            acc[t] = __builtin_amdgcn_wmma_f32_16x16x32_f16(
                false, a, false, b, (short)0, acc[t], false, false);
        }
    }

    // --- epilogue: C/D layout (VGPR i -> row i / i+8 by lane half; col=lane&15)
#pragma unroll
    for (int t = 0; t < 4; ++t) {
        const int col = colbase + t * 16 + lr;
        const float bv = bias ? bias[col] : 0.0f;
#pragma unroll
        for (int i = 0; i < 8; ++i) {
            const int rr   = rbase + i + (hi << 3);
            const int orow = strided ? ((rr / mcount) * MTOT + mstart + (rr % mcount)) : rr;
            const size_t idx = (size_t)orow * Cout + col;
            const float val = acc[t][i] + bv;
            if (accum) Out[idx] += val; else Out[idx] = val;
        }
    }
}

// ---------------------------------------------------------------------------
// Edge preprocessing: envelope, radial basis scatter, SH scatter, env store
// ---------------------------------------------------------------------------
__global__ void edge_prep_kernel(const int* __restrict__ ei, const int* __restrict__ z,
                                 const float* __restrict__ dist, const float* __restrict__ dvec,
                                 const float* __restrict__ freq, const float* __restrict__ mul,
                                 const float* __restrict__ biasE,
                                 float* __restrict__ summed, float* __restrict__ eq9,
                                 float* __restrict__ envOut)
{
    int e = blockIdx.x * blockDim.x + threadIdx.x;
    if (e >= NEDGES) return;
    const int src = ei[e], dst = ei[NEDGES + e];
    const float d  = dist[e];
    const float r  = d * (1.0f / RCUT);
    const float r2 = r * r, r3 = r2 * r;
    float env = 1.0f - 6.0f * r3 * r2 + 15.0f * r2 * r2 - 10.0f * r3;
    env = fmaxf(env, 0.0f);
    envOut[e] = env;

    const int   et = z[src] * MAXZ + z[dst];
    const float mm = mul[et], be = biasE[et];
    const float inv_xr = 1.0f / r;
    for (int b = 0; b < NBASIS; ++b) {
        float bas = NORMC * inv_xr * __sinf(freq[b] * r);
        bas = (bas * mm + be) * env;
        atomicAdd(&summed[(size_t)dst * NBASIS + b], bas);
    }

    const float vx = dvec[e * 3 + 0], vy = dvec[e * 3 + 1], vz = dvec[e * 3 + 2];
    const float nrm = sqrtf(vx * vx + vy * vy + vz * vz) + 1e-12f;
    const float x = vx / nrm, y = vy / nrm, zz = vz / nrm;
    const float s3 = 1.7320508075688772f, s5 = 2.23606797749979f, s15 = 3.872983346207417f;
    float sh[MTOT];
    sh[0] = 1.0f;        sh[1] = s3 * y;       sh[2] = s3 * zz;      sh[3] = s3 * x;
    sh[4] = s15 * x * y; sh[5] = s15 * y * zz; sh[6] = 0.5f * s5 * (3.0f * zz * zz - 1.0f);
    sh[7] = s15 * x * zz; sh[8] = 0.5f * s15 * (x * x - y * y);
    for (int m = 0; m < MTOT; ++m)
        atomicAdd(&eq9[(size_t)dst * MTOT + m], sh[m] * env);
}

// ---------------------------------------------------------------------------
// Node init: gating of inv/equ -> node features x[N,9,64]
// ---------------------------------------------------------------------------
__global__ void node_init_kernel(const int* __restrict__ z, const float* __restrict__ emb,
                                 const float* __restrict__ inv_in, const float* __restrict__ eq9,
                                 const float* __restrict__ eqw, const float* __restrict__ eqb,
                                 float* __restrict__ x)
{
    int t = blockIdx.x * blockDim.x + threadIdx.x;
    if (t >= NNODES * CDIM) return;
    const int n = t >> 6, c = t & 63;
    const float* e = emb + (size_t)z[n] * (2 * CDIM);
    const float g0 = sigmoidf_(inv_in[(size_t)n * 128 + c]       + e[c]);
    const float g1 = sigmoidf_(inv_in[(size_t)n * 128 + 64 + c]  + e[64 + c]);
    const float* q9 = eq9 + (size_t)n * MTOT;
    float* xr = x + (size_t)n * MTOT * CDIM + c;
    const float e0 = q9[0] * eqw[c] + eqb[c];
    xr[0] = e0 * sigmoidf_(e0);                          // silu
    for (int m = 1; m < 4; ++m) xr[m * CDIM] = q9[m] * eqw[64 + c]  * g0;
    for (int m = 4; m < 9; ++m) xr[m * CDIM] = q9[m] * eqw[128 + c] * g1;
}

// ---------------------------------------------------------------------------
// Per-degree equivariant RMS norm. One block (64 thr) per node.
// ---------------------------------------------------------------------------
__global__ void rmsnorm_kernel(const float* __restrict__ x, const float* __restrict__ w,
                               float* __restrict__ h)
{
    __shared__ float r0[64], r1[64], r2[64];
    const int n = blockIdx.x, c = threadIdx.x;
    const float* xr = x + (size_t)n * MTOT * CDIM + c;
    float s0, s1 = 0.0f, s2 = 0.0f;
    { float v = xr[0]; s0 = v * v; }
    for (int m = 1; m < 4; ++m) { float v = xr[m * CDIM]; s1 += v * v; }
    for (int m = 4; m < 9; ++m) { float v = xr[m * CDIM]; s2 += v * v; }
    r0[c] = s0; r1[c] = s1; r2[c] = s2;
    __syncthreads();
    for (int off = 32; off > 0; off >>= 1) {
        if (c < off) { r0[c] += r0[c + off]; r1[c] += r1[c + off]; r2[c] += r2[c + off]; }
        __syncthreads();
    }
    const float rs0 = rsqrtf(r0[0] * (1.0f / (1.0f * CDIM)) + 1e-7f);
    const float rs1 = rsqrtf(r1[0] * (1.0f / (3.0f * CDIM)) + 1e-7f);
    const float rs2 = rsqrtf(r2[0] * (1.0f / (5.0f * CDIM)) + 1e-7f);
    float* hr = h + (size_t)n * MTOT * CDIM + c;
    hr[0] = xr[0] * rs0 * w[c];
    for (int m = 1; m < 4; ++m) hr[m * CDIM] = xr[m * CDIM] * rs1 * w[64 + c];
    for (int m = 4; m < 9; ++m) hr[m * CDIM] = xr[m * CDIM] * rs2 * w[128 + c];
}

// ---------------------------------------------------------------------------
// Attention passes
// ---------------------------------------------------------------------------
__global__ void attn_logits_kernel(const int* __restrict__ ei, const float* __restrict__ q,
                                   const float* __restrict__ k, float* __restrict__ logits,
                                   float* __restrict__ mx)
{
    int t = blockIdx.x * blockDim.x + threadIdx.x;
    if (t >= NEDGES * NHEADS) return;
    const int e = t >> 3, hh = t & 7;
    const int src = ei[e], dst = ei[NEDGES + e];
    const float* qe = q + (size_t)dst * (MTOT * CDIM) + hh * DHEAD;
    const float* ke = k + (size_t)src * (MTOT * CDIM) + hh * DHEAD;
    float s = 0.0f;
    for (int m = 0; m < MTOT; ++m)
        for (int dd = 0; dd < DHEAD; ++dd)
            s += qe[m * CDIM + dd] * ke[m * CDIM + dd];
    s = s * 0.11785113019775792f /* 1/sqrt(72) */ + 1.0f /* attn_bias */;
    logits[t] = s;
    atomicMaxF(&mx[dst * NHEADS + hh], s);
}

__global__ void attn_exp_kernel(const int* __restrict__ ei, float* __restrict__ logits,
                                const float* __restrict__ mx, float* __restrict__ den)
{
    int t = blockIdx.x * blockDim.x + threadIdx.x;
    if (t >= NEDGES * NHEADS) return;
    const int e = t >> 3, hh = t & 7;
    const int dst = ei[NEDGES + e];
    const float ex = __expf(logits[t] - mx[dst * NHEADS + hh]);
    logits[t] = ex;
    atomicAdd(&den[dst * NHEADS + hh], ex);
}

__global__ void attn_accum_kernel(const int* __restrict__ ei, const float* __restrict__ ex,
                                  const float* __restrict__ den, const float* __restrict__ env,
                                  const float* __restrict__ v, float* __restrict__ outAcc)
{
    long long t = (long long)blockIdx.x * blockDim.x + threadIdx.x;
    if (t >= (long long)NEDGES * MTOT * CDIM) return;
    const int e  = (int)(t / (MTOT * CDIM));
    const int mc = (int)(t % (MTOT * CDIM));
    const int hh = (mc & 63) >> 3;
    const int src = ei[e], dst = ei[NEDGES + e];
    const float alpha = ex[e * NHEADS + hh] / (den[dst * NHEADS + hh] + 1e-9f) * env[e];
    atomicAdd(&outAcc[(size_t)dst * (MTOT * CDIM) + mc], alpha * v[(size_t)src * (MTOT * CDIM) + mc]);
}

// ---------------------------------------------------------------------------
// FFN gate: gate = sigmoid(h[:, 0]); row0 -> silu (== h0*gate); rows1..8 *= gate
// ---------------------------------------------------------------------------
__global__ void ffn_gate_kernel(float* __restrict__ f)
{
    int t = blockIdx.x * blockDim.x + threadIdx.x;
    if (t >= NNODES * 4 * CDIM) return;
    const int n = t >> 8, c = t & 255;
    float* fr = f + (size_t)n * MTOT * (4 * CDIM) + c;
    const float g = sigmoidf_(fr[0]);
    for (int m = 0; m < MTOT; ++m) fr[m * 4 * CDIM] *= g;
}

// ---------------------------------------------------------------------------
// Host-side launch helpers
// ---------------------------------------------------------------------------
static inline void launch_fill(hipStream_t s, float* p, float v, size_t n) {
    fill_kernel<<<dim3((unsigned)((n + 255) / 256)), 256, 0, s>>>(p, v, (int)n);
}
static inline void launch_swz(hipStream_t s, const float* src, _Float16* dst,
                              int Cin, int Cout, int nmat) {
    const int n = nmat * Cin * Cout;
    swizzle_w_kernel<<<dim3((n + 255) / 256), 256, 0, s>>>(src, dst, Cin, Cout, nmat);
}
static inline void so3_gemm(hipStream_t s, const float* A, const _Float16* W, const float* bias,
                            float* Out, int rows, int Cin, int Cout,
                            int mstart, int mcount, int strided, int accum) {
    dim3 g(rows / 64, Cout / 64);
    if (Cin == 64)
        so3_gemm_kernel<64><<<g, 128, 0, s>>>(A, W, bias, Out, Cout, mstart, mcount, strided, accum);
    else
        so3_gemm_kernel<256><<<g, 128, 0, s>>>(A, W, bias, Out, Cout, mstart, mcount, strided, accum);
}

extern "C" void kernel_launch(void* const* d_in, const int* in_sizes, int n_in,
                              void* d_out, int out_size, void* d_ws, size_t ws_size,
                              hipStream_t stream)
{
    (void)in_sizes; (void)n_in; (void)out_size; (void)ws_size;
    const int*   edge_index = (const int*)d_in[0];
    const int*   z          = (const int*)d_in[1];
    const float* dist       = (const float*)d_in[2];
    const float* dvec       = (const float*)d_in[3];
    const float* freq       = (const float*)d_in[4];
    const float* mul        = (const float*)d_in[5];
    const float* biasE      = (const float*)d_in[6];
    const float* emb        = (const float*)d_in[7];
    const float* radW       = (const float*)d_in[8];
    const float* radB       = (const float*)d_in[9];
    const float* eqw        = (const float*)d_in[10];
    const float* eqb        = (const float*)d_in[11];

    float* wsf = (float*)d_ws;
    float* X   = wsf + OFF_X;
    float* Hh  = wsf + OFF_H;
    float* Q   = wsf + OFF_Q;
    float* K   = wsf + OFF_K;
    float* V   = wsf + OFF_V;
    float* FFN = wsf + OFF_FFN;
    float* LOG = wsf + OFF_LOG;
    float* MX  = wsf + OFF_MX;
    float* DEN = wsf + OFF_DEN;
    float* ENV = wsf + OFF_ENV;
    float* SUM = wsf + OFF_SUM;
    float* EQ9 = wsf + OFF_EQ9;
    float* INV = wsf + OFF_INV;
    _Float16* arena = (_Float16*)(wsf + FLT_END);

    // --- swizzle GEMM weights into f16 fragment-order arena ---
    launch_swz(stream, radW, arena + AOFF_RAD, 64, 128, 1);
    for (int b = 0; b < 2; ++b) {
        void* const* bp = d_in + 12 + 14 * b;
        _Float16* ab = arena + AOFF_BLK(b);
        launch_swz(stream, (const float*)bp[2],  ab + AW_Q, 64, 64, 3);    // wq
        launch_swz(stream, (const float*)bp[4],  ab + AW_K, 64, 64, 3);    // wk
        launch_swz(stream, (const float*)bp[6],  ab + AW_V, 64, 64, 3);    // wv
        launch_swz(stream, (const float*)bp[8],  ab + AW_O, 64, 64, 3);    // wo
        launch_swz(stream, (const float*)bp[10], ab + AW_1, 64, 256, 3);   // w1
        launch_swz(stream, (const float*)bp[12], ab + AW_2, 256, 64, 3);   // w2
    }

    // --- edge scatter ---
    launch_fill(stream, SUM, 0.0f, (size_t)NNODES * NBASIS);
    launch_fill(stream, EQ9, 0.0f, (size_t)NNODES * MTOT);
    edge_prep_kernel<<<dim3((NEDGES + 255) / 256), 256, 0, stream>>>(
        edge_index, z, dist, dvec, freq, mul, biasE, SUM, EQ9, ENV);

    // --- radial projection: inv = summed @ radial_W + radial_b (WMMA) ---
    so3_gemm(stream, SUM, arena + AOFF_RAD, radB, INV, NNODES, 64, 128, 0, 1, 0, 0);

    // --- node features ---
    node_init_kernel<<<dim3((NNODES * CDIM + 255) / 256), 256, 0, stream>>>(
        z, emb, INV, EQ9, eqw, eqb, X);

    static const int DS[3] = {0, 1, 4};
    static const int DC[3] = {1, 3, 5};

    for (int b = 0; b < 2; ++b) {
        void* const* bp = d_in + 12 + 14 * b;
        const float* norm1 = (const float*)bp[0];
        const float* norm2 = (const float*)bp[1];
        const float* bq = (const float*)bp[3];
        const float* bk = (const float*)bp[5];
        const float* bv = (const float*)bp[7];
        const float* bo = (const float*)bp[9];
        const float* b1 = (const float*)bp[11];
        const float* b2 = (const float*)bp[13];
        _Float16* ab = arena + AOFF_BLK(b);

        // ---- attention ----
        rmsnorm_kernel<<<dim3(NNODES), 64, 0, stream>>>(X, norm1, Hh);
        for (int l = 0; l < 3; ++l) {
            const int rows = NNODES * DC[l];
            so3_gemm(stream, Hh, ab + AW_Q + l * 64 * 64, l == 0 ? bq : nullptr, Q, rows, 64, 64, DS[l], DC[l], 1, 0);
            so3_gemm(stream, Hh, ab + AW_K + l * 64 * 64, l == 0 ? bk : nullptr, K, rows, 64, 64, DS[l], DC[l], 1, 0);
            so3_gemm(stream, Hh, ab + AW_V + l * 64 * 64, l == 0 ? bv : nullptr, V, rows, 64, 64, DS[l], DC[l], 1, 0);
        }
        launch_fill(stream, MX, -3.0e38f, (size_t)NNODES * NHEADS);
        launch_fill(stream, DEN, 0.0f, (size_t)NNODES * NHEADS);
        attn_logits_kernel<<<dim3((NEDGES * NHEADS + 255) / 256), 256, 0, stream>>>(
            edge_index, Q, K, LOG, MX);
        attn_exp_kernel<<<dim3((NEDGES * NHEADS + 255) / 256), 256, 0, stream>>>(
            edge_index, LOG, MX, DEN);
        launch_fill(stream, Hh, 0.0f, NMC);   // Hh reused as attention accumulator
        {
            long long tot = (long long)NEDGES * MTOT * CDIM;
            attn_accum_kernel<<<dim3((unsigned)((tot + 255) / 256)), 256, 0, stream>>>(
                edge_index, LOG, DEN, ENV, V, Hh);
        }
        for (int l = 0; l < 3; ++l)   // x += so3_linear(attn_out, wo, bo)
            so3_gemm(stream, Hh, ab + AW_O + l * 64 * 64, l == 0 ? bo : nullptr, X,
                     NNODES * DC[l], 64, 64, DS[l], DC[l], 1, 1);

        // ---- FFN ----
        rmsnorm_kernel<<<dim3(NNODES), 64, 0, stream>>>(X, norm2, Hh);
        for (int l = 0; l < 3; ++l)
            so3_gemm(stream, Hh, ab + AW_1 + l * 64 * 256, l == 0 ? b1 : nullptr, FFN,
                     NNODES * DC[l], 64, 256, DS[l], DC[l], 1, 0);
        ffn_gate_kernel<<<dim3((NNODES * 4 * CDIM + 255) / 256), 256, 0, stream>>>(FFN);
        for (int l = 0; l < 3; ++l)   // x += so3_linear(h, w2, b2)
            so3_gemm(stream, FFN, ab + AW_2 + l * 256 * 64, l == 0 ? b2 : nullptr, X,
                     NNODES * DC[l], 256, 64, DS[l], DC[l], 1, 1);
    }

    copy_kernel<<<dim3((unsigned)((NMC + 255) / 256)), 256, 0, stream>>>(X, (float*)d_out, (int)NMC);
}
